// VectorQuantizer_12945031431013
// MI455X (gfx1250) — matrix operand order repaired and checked by reading, not compiled, not run on hardware
//
#include <hip/hip_runtime.h>
#include <hip/hip_bf16.h>

// ---------------- problem constants ----------------
#define NTOK   65536
#define DIM    64
#define KCODES 8192
#define DECAY  0.99f
#define EPSV   1e-05f

// output layout (flat concat, reference return order, all float32;
// indices written as float(index) in the final kernel)
#define ZQ_OFF   0u
#define IDX_OFF  4194304u              // NTOK*DIM
#define LOSS_OFF 4259840u              // + NTOK
#define CB_OFF   4259841u              // + 1
#define CS_OFF   4784129u              // + KCODES*DIM
#define EMA_OFF  4792321u              // + KCODES

// workspace layout (floats): [0]=n_total, [1]=loss_sum, counts @16, embed_sum @16+K
#define WS_COUNTS 16u
#define WS_EMBED  (16u + KCODES)
#define WS_ZERO_N (16u + KCODES + KCODES * DIM)

typedef float  v8f  __attribute__((ext_vector_type(8)));
typedef __bf16 v8bf __attribute__((ext_vector_type(8)));
typedef __bf16 v16bf __attribute__((ext_vector_type(16)));

#define TILE_K 128     // codes staged in LDS per iteration
#define ROWP   72      // padded bf16 row stride (72*2=144B -> conflict-free)
#define NTILES (KCODES / TILE_K)

__device__ __forceinline__ v16bf load16bf(const __bf16* p) {
  // two 16B-aligned 8-elem chunks at p and p+16 -> one v16bf fragment (ds_load_b128 x2)
  v8bf t0 = *(const v8bf*)p;
  v8bf t1 = *(const v8bf*)(p + 16);
  v16bf r;
#pragma unroll
  for (int e = 0; e < 8; ++e) { r[e] = t0[e]; r[8 + e] = t1[e]; }
  return r;
}

// convert a register-staged half row (32 f32) to bf16 hi/lo planes in LDS,
// and produce e2 for the row via a lane-pair shuffle (no barrier needed).
__device__ __forceinline__ void stage_store(const float4* stg,
                                            __bf16* __restrict__ hd_base,
                                            __bf16* __restrict__ ld_base,
                                            float* __restrict__ e2_base,
                                            int r, int h) {
  __bf16* hd = hd_base + r * ROWP + h * 32;
  __bf16* ld = ld_base + r * ROWP + h * 32;
  float p = 0.f;
#pragma unroll
  for (int i = 0; i < 8; ++i) {
    const float xs[4] = {stg[i].x, stg[i].y, stg[i].z, stg[i].w};
#pragma unroll
    for (int j = 0; j < 4; ++j) {
      float x = xs[j];
      __bf16 xh = (__bf16)x;
      hd[4 * i + j] = xh;
      ld[4 * i + j] = (__bf16)(x - (float)xh);
      p += x * x;
    }
  }
  p += __shfl_xor(p, 1, 32);        // partner holds the other half-row
  if (h == 0) e2_base[r] = p;
}

// ---------------- kernel 0: zero workspace ----------------
__global__ __launch_bounds__(256) void vq_zero(float* __restrict__ ws) {
  unsigned i = blockIdx.x * 256u + threadIdx.x;
  if (i < WS_ZERO_N) ws[i] = 0.0f;
}

// ---------------- kernel 1: WMMA argmin over codes ----------------
// block = 256 threads = 8 waves; each wave owns 16 tokens vs all 8192 codes.
// Double-buffered LDS staging: global loads for tile t+1 are issued into
// registers BEFORE compute of tile t, converted/stored after -> HBM latency
// fully hidden behind 48 WMMAs per tile. One barrier per tile.
__global__ __launch_bounds__(256) void vq_argmin(
    const float* __restrict__ z, const float* __restrict__ codebook,
    int* __restrict__ idx_out) {
  __shared__ __align__(16) __bf16 lds_hi[2][TILE_K * ROWP];
  __shared__ __align__(16) __bf16 lds_lo[2][TILE_K * ROWP];
  __shared__ float lds_e2[2][TILE_K];

  const int tid   = threadIdx.x;
  const int lane  = tid & 31;
  const int wave  = tid >> 5;
  const int khalf = lane >> 4;     // which 16-lane half
  const int lrow  = lane & 15;
  const int srow_r = tid >> 1;     // staging: row per thread-pair
  const int srow_h = tid & 1;      // staging: which half-row

  const int wave_tok = blockIdx.x * 128 + wave * 16;

  // ---- A fragments: 16 tokens x D=64, split bf16 hi/lo, two K=32 steps ----
  // 16-bit A 16x32 layout: lanes0-15 hold K=[0..7]+[16..23], lanes16-31 +8.
  const float* zrow = z + (size_t)(wave_tok + lrow) * DIM;
  v16bf ah[2], al[2];
#pragma unroll
  for (int s = 0; s < 2; ++s) {
    const int b0 = 32 * s + 8 * khalf;        // elements 0..7  <- K contiguous
    const int b1 = 32 * s + 16 + 8 * khalf;   // elements 8..15 <- K contiguous
#pragma unroll
    for (int e = 0; e < 8; ++e) {
      float x0 = zrow[b0 + e], x1 = zrow[b1 + e];
      __bf16 h0 = (__bf16)x0, h1 = (__bf16)x1;
      ah[s][e] = h0;        ah[s][8 + e] = h1;
      al[s][e] = (__bf16)(x0 - (float)h0);
      al[s][8 + e] = (__bf16)(x1 - (float)h1);
    }
  }

  float minv[8];
  int   mini[8];
#pragma unroll
  for (int v = 0; v < 8; ++v) { minv[v] = 3.4e38f; mini[v] = 0; }

  // ---- prologue: stage tile 0 ----
  {
    const float* srow = codebook + (size_t)srow_r * DIM + srow_h * 32;
    float4 stg0[8];
#pragma unroll
    for (int i = 0; i < 8; ++i) stg0[i] = *(const float4*)(srow + 4 * i);
    stage_store(stg0, lds_hi[0], lds_lo[0], lds_e2[0], srow_r, srow_h);
  }
  __syncthreads();

#pragma unroll 1
  for (int t = 0; t < NTILES; ++t) {
    const int cur = t & 1;

    // ---- issue global loads for tile t+1 (no wait until after compute) ----
    float4 stg[8];
    if (t + 1 < NTILES) {
      const float* srow =
          codebook + ((size_t)(t + 1) * TILE_K + srow_r) * DIM + srow_h * 32;
#pragma unroll
      for (int i = 0; i < 8; ++i) stg[i] = *(const float4*)(srow + 4 * i);
      if (t + 2 < NTILES)   // pull t+2 toward L2
        __builtin_prefetch(codebook + (size_t)(t + 2) * TILE_K * DIM + tid * 32,
                           0, 0);
    }

    // ---- compute: 8 subtiles of 16 codes; 6 WMMAs each (split-bf16) ----
    const __bf16* hbase = lds_hi[cur];
    const __bf16* lbase = lds_lo[cur];
    const float*  e2b   = lds_e2[cur];
#pragma unroll 1
    for (int sub = 0; sub < 8; ++sub) {
      const int nloc = sub * 16 + lrow;       // B col N = lane&15
      const __bf16* hb = hbase + nloc * ROWP;
      const __bf16* lb = lbase + nloc * ROWP;
      v8f acc0 = {};   // hi*hi chain (2 deep)
      v8f acc1 = {};   // cross-term chain (4 deep)
#pragma unroll
      for (int s = 0; s < 2; ++s) {
        // B 32x16 layout: lanes0-15 hold K=[0..7]+[16..23], lanes16-31 +8
        const int off = 32 * s + 8 * khalf;
        v16bf bh = load16bf(hb + off);
        v16bf bl = load16bf(lb + off);
        acc0 = __builtin_amdgcn_wmma_f32_16x16x32_bf16(
            false, ah[s], false, bh, (short)0, acc0, false, false);
        acc1 = __builtin_amdgcn_wmma_f32_16x16x32_bf16(
            false, al[s], false, bh, (short)0, acc1, false, false);
        acc1 = __builtin_amdgcn_wmma_f32_16x16x32_bf16(
            false, ah[s], false, bl, (short)0, acc1, false, false);
      }
      const float e2v = e2b[nloc];
      const int  code = t * TILE_K + nloc;
#pragma unroll
      for (int v = 0; v < 8; ++v) {
        float dist = e2v - 2.0f * (acc0[v] + acc1[v]); // |z|^2 const per token
        if (dist < minv[v]) { minv[v] = dist; mini[v] = code; }
      }
    }

    // ---- convert + store tile t+1 into the back buffer ----
    if (t + 1 < NTILES)
      stage_store(stg, lds_hi[cur ^ 1], lds_lo[cur ^ 1], lds_e2[cur ^ 1],
                  srow_r, srow_h);
    __syncthreads();
  }

  // ---- reduce (min,argmin) across the 16 lanes of each half ----
#pragma unroll
  for (int m = 1; m < 16; m <<= 1) {
#pragma unroll
    for (int v = 0; v < 8; ++v) {
      float ov = __shfl_xor(minv[v], m, 32);
      int   oi = __shfl_xor(mini[v], m, 32);
      if (ov < minv[v] || (ov == minv[v] && oi < mini[v])) {
        minv[v] = ov; mini[v] = oi;
      }
    }
  }
  if (lrow == 0) {                 // element v of half khalf = token v+8*khalf
#pragma unroll
    for (int v = 0; v < 8; ++v)
      idx_out[wave_tok + v + 8 * khalf] = mini[v];
  }
}

// ---------------- kernel 2: gather z_q, commit loss, segment sums ----------------
__global__ __launch_bounds__(256) void vq_scatter(
    const float* __restrict__ z, const float* __restrict__ codebook,
    const int* __restrict__ idx, float* __restrict__ zq_out,
    float* __restrict__ counts, float* __restrict__ embed_sum,
    float* __restrict__ loss_accum) {
  const int tok = blockIdx.x * 256 + threadIdx.x;
  float lsum = 0.f;
  const int k = idx[tok];
  const float* e  = codebook + (size_t)k * DIM;
  const float* zr = z + (size_t)tok * DIM;
  float* o = zq_out + (size_t)tok * DIM;
#pragma unroll
  for (int d = 0; d < DIM; d += 4) {
    float4 ev = *(const float4*)(e + d);
    float4 zv = *(const float4*)(zr + d);
    *(float4*)(o + d) = ev;                            // z_q_st forward == z_q
    float dx = zv.x - ev.x, dy = zv.y - ev.y;
    float dz = zv.z - ev.z, dw = zv.w - ev.w;
    lsum += dx * dx + dy * dy + dz * dz + dw * dw;
    atomicAdd(&embed_sum[(size_t)k * DIM + d + 0], zv.x);
    atomicAdd(&embed_sum[(size_t)k * DIM + d + 1], zv.y);
    atomicAdd(&embed_sum[(size_t)k * DIM + d + 2], zv.z);
    atomicAdd(&embed_sum[(size_t)k * DIM + d + 3], zv.w);
  }
  atomicAdd(&counts[k], 1.0f);

  __shared__ float red[256];
  red[threadIdx.x] = lsum;
  __syncthreads();
  for (int s = 128; s > 0; s >>= 1) {
    if (threadIdx.x < s) red[threadIdx.x] += red[threadIdx.x + s];
    __syncthreads();
  }
  if (threadIdx.x == 0) atomicAdd(loss_accum, red[0]);
}

// ---------------- kernel 3: EMA cluster_size + total n ----------------
__global__ __launch_bounds__(256) void vq_ema_cs(
    const float* __restrict__ cluster_size, const float* __restrict__ counts,
    float* __restrict__ out_cs, float* __restrict__ n_accum) {
  const int k = blockIdx.x * 256 + threadIdx.x;
  float ncs = cluster_size[k] * DECAY + counts[k] * (1.0f - DECAY);
  out_cs[k] = ncs;
  __shared__ float red[256];
  red[threadIdx.x] = ncs;
  __syncthreads();
  for (int s = 128; s > 0; s >>= 1) {
    if (threadIdx.x < s) red[threadIdx.x] += red[threadIdx.x + s];
    __syncthreads();
  }
  if (threadIdx.x == 0) atomicAdd(n_accum, red[0]);
}

// ---------------- kernel 4: new_ema, new_codebook, loss, idx->float ----------------
__global__ __launch_bounds__(256) void vq_final(
    const float* __restrict__ ema_weights, const float* __restrict__ embed_sum,
    const float* __restrict__ out_cs, const float* __restrict__ ws,
    float* __restrict__ out_cb, float* __restrict__ out_ema,
    float* __restrict__ out_loss, float* __restrict__ out_idx_f) {
  const unsigned i = blockIdx.x * 256u + threadIdx.x;
  const float n = ws[0];
  if (i < KCODES * DIM) {
    const int k = i / DIM;
    float ne = ema_weights[i] * DECAY + embed_sum[i] * (1.0f - DECAY);
    out_ema[i] = ne;
    float smoothed = (out_cs[k] + EPSV) / (n + (float)KCODES * EPSV) * n;
    out_cb[i] = ne / smoothed;
  }
  if (i < NTOK) {
    int v = ((const int*)out_idx_f)[i];   // argmin kernel wrote raw int32
    out_idx_f[i] = (float)v;              // reference output as float
  }
  if (i == 0) out_loss[0] = ws[1] / (float)(NTOK * DIM);
}

// ---------------- launcher ----------------
extern "C" void kernel_launch(void* const* d_in, const int* in_sizes, int n_in,
                              void* d_out, int out_size, void* d_ws, size_t ws_size,
                              hipStream_t stream) {
  (void)in_sizes; (void)n_in; (void)out_size; (void)ws_size;
  const float* z            = (const float*)d_in[0];
  const float* codebook     = (const float*)d_in[1];
  const float* cluster_size = (const float*)d_in[2];
  const float* ema_weights  = (const float*)d_in[3];

  float* out      = (float*)d_out;
  float* out_zq   = out + ZQ_OFF;
  int*   out_idxi = (int*)(out + IDX_OFF);
  float* out_loss = out + LOSS_OFF;
  float* out_cb   = out + CB_OFF;
  float* out_cs   = out + CS_OFF;
  float* out_ema  = out + EMA_OFF;

  float* ws        = (float*)d_ws;
  float* ws_counts = ws + WS_COUNTS;
  float* ws_embed  = ws + WS_EMBED;

  vq_zero<<<(WS_ZERO_N + 255) / 256, 256, 0, stream>>>(ws);
  vq_argmin<<<NTOK / 128, 256, 0, stream>>>(z, codebook, out_idxi);
  vq_scatter<<<NTOK / 256, 256, 0, stream>>>(z, codebook, out_idxi, out_zq,
                                             ws_counts, ws_embed, ws + 1);
  vq_ema_cs<<<KCODES / 256, 256, 0, stream>>>(cluster_size, ws_counts, out_cs, ws);
  vq_final<<<(KCODES * DIM + 255) / 256, 256, 0, stream>>>(
      ema_weights, ws_embed, out_cs, ws, out_cb, out_ema, out_loss,
      (float*)out_idxi);
}
